// GridToSequence_60395830117199
// MI455X (gfx1250) — matrix operand order repaired and checked by reading, not compiled, not run on hardware
//
#include <hip/hip_runtime.h>

// ---------------- problem constants ----------------
#define BD   8
#define TD   576
#define LD   1024
#define DD   1024
#define POSD 512
#define HD   16
#define HSD  64
#define DFFD 4096

typedef __attribute__((ext_vector_type(16))) __bf16 v16bf;
typedef __attribute__((ext_vector_type(8)))  float  v8f;
typedef __attribute__((ext_vector_type(4)))  int    v4i;

union FragAB { unsigned int u[8]; v16bf v; };
union FragC  { float f[8]; v8f v; };

#if __has_builtin(__builtin_amdgcn_global_load_async_to_lds_b128) && __has_builtin(__builtin_amdgcn_s_wait_asynccnt)
#define USE_ASYNC_LDS 1
#else
#define USE_ASYNC_LDS 0
#endif

#if USE_ASYNC_LDS
// one 16-byte cache->LDS copy, no VGPR data path, tracked by ASYNCcnt.
__device__ __forceinline__ void async_copy16(const unsigned short* g, unsigned short* l) {
    __builtin_amdgcn_global_load_async_to_lds_b128(
        (__attribute__((address_space(1))) v4i*)g,
        (__attribute__((address_space(3))) v4i*)l, 0, 0);
}
#endif

__device__ __forceinline__ unsigned short f2bf(float f) {
    unsigned int u = __float_as_uint(f);
    u += 0x7FFFu + ((u >> 16) & 1u);
    return (unsigned short)(u >> 16);
}
__device__ __forceinline__ v8f zero8() {
    FragC c;
#pragma unroll
    for (int r = 0; r < 8; ++r) c.f[r] = 0.f;
    return c.v;
}
__device__ __forceinline__ v8f wmma_bf16(v16bf a, v16bf b, v8f c) {
    return __builtin_amdgcn_wmma_f32_16x16x32_bf16(false, a, false, b, (short)0, c, false, false);
}

// A-matrix fragment (16x32 bf16) from row-major [m][k] LDS tile.
// Layout: lanes0-15 m=lane, v0-7 -> k pairs 0..15;
//         lanes16-31 m=lane-16, v0-7 -> k pairs 16..31.
__device__ __forceinline__ v16bf lds_a_frag(const unsigned short* base, int stride) {
    int lane = threadIdx.x & 31;
    int m = lane & 15, lh = lane >> 4;
    FragAB f;
#pragma unroll
    for (int v = 0; v < 8; ++v) {
        int kp = 2 * v + (lh ? 16 : 0);
        f.u[v] = *(const unsigned int*)(base + m * stride + kp);
    }
    return f.v;
}
// Same mapping from global memory (Q fragments in attention).
__device__ __forceinline__ v16bf glb_a_frag(const unsigned short* base, long stride, int kbase) {
    int lane = threadIdx.x & 31;
    int m = lane & 15, lh = lane >> 4;
    FragAB f;
#pragma unroll
    for (int v = 0; v < 8; ++v) {
        int kp = 2 * v + (lh ? 16 : 0);
        f.u[v] = *(const unsigned int*)(base + (long)m * stride + kbase + kp);
    }
    return f.v;
}
// B-matrix fragment (32x16) from an [n][k]-layout LDS tile (k contiguous).
// lanes0-15: n=lane, k=0..15; lanes16-31: n=lane-16, k=16..31.
__device__ __forceinline__ v16bf lds_b_frag_nk(const unsigned short* base, int stride, int kbase) {
    int lane = threadIdx.x & 31;
    int n = lane & 15, lh = lane >> 4;
    FragAB f;
#pragma unroll
    for (int v = 0; v < 8; ++v) {
        int k = kbase + (lh ? 16 : 0) + 2 * v;
        f.u[v] = *(const unsigned int*)(base + n * stride + k);
    }
    return f.v;
}
// B-matrix fragment from a raw row-major [k][n] LDS tile (strided u16 reads;
// 16 consecutive lanes read 32 consecutive bytes -> conflict free).
__device__ __forceinline__ v16bf lds_b_frag_kn(const unsigned short* base, int stride, int kbase) {
    int lane = threadIdx.x & 31;
    int n = lane & 15, lh = lane >> 4;
    FragAB f;
#pragma unroll
    for (int v = 0; v < 8; ++v) {
        int k = kbase + (lh ? 16 : 0) + 2 * v;
        unsigned int lo = base[k * stride + n];
        unsigned int hi = base[(k + 1) * stride + n];
        f.u[v] = lo | (hi << 16);
    }
    return f.v;
}

// ---------------- elementwise / layernorm ----------------
__global__ void cvt_f32_bf16(const float* __restrict__ src, unsigned short* __restrict__ dst, int n) {
    int i = blockIdx.x * blockDim.x + threadIdx.x;
    if (i < n) dst[i] = f2bf(src[i]);
}

__global__ __launch_bounds__(256) void layernorm_bf16(
    const float* __restrict__ x, const float* __restrict__ g, const float* __restrict__ b,
    unsigned short* __restrict__ out, int width) {
    long row = blockIdx.x;
    const float* xr = x + row * width;
    __shared__ float red[256];
    float s = 0.f, s2 = 0.f;
    for (int i = threadIdx.x; i < width; i += 256) { float v = xr[i]; s += v; s2 += v * v; }
    red[threadIdx.x] = s; __syncthreads();
    for (int o = 128; o > 0; o >>= 1) { if ((int)threadIdx.x < o) red[threadIdx.x] += red[threadIdx.x + o]; __syncthreads(); }
    float mean = red[0] / width; __syncthreads();
    red[threadIdx.x] = s2; __syncthreads();
    for (int o = 128; o > 0; o >>= 1) { if ((int)threadIdx.x < o) red[threadIdx.x] += red[threadIdx.x + o]; __syncthreads(); }
    float var = red[0] / width - mean * mean;
    float rs = rsqrtf(var + 1e-5f);
    unsigned short* orow = out + row * width;
    for (int i = threadIdx.x; i < width; i += 256)
        orow[i] = f2bf((xr[i] - mean) * rs * g[i] + b[i]);
}

// ---------------- generic bf16 WMMA GEMM ----------------
// 256 threads = 8 waves (2x4); WG tile 64(M) x 128(N); wave tile 32x32; K chunks of 32.
// Double-buffered LDS; async copies for chunk kc+1 stay in flight behind the
// WMMA work on chunk kc (staggered s_wait_asynccnt(3): 3 copies/thread/chunk).
template<bool BIAS, bool GELU, bool RES, bool OUT_F32, bool OUT_BF16>
__global__ __launch_bounds__(256) void gemm_bf16(
    const unsigned short* __restrict__ A, const unsigned short* __restrict__ Bm,
    const float* __restrict__ bias, const float* __restrict__ res,
    float* __restrict__ outF, unsigned short* __restrict__ outH,
    int M, int N, int K) {
    __shared__ unsigned short As[2][64][32];
    __shared__ unsigned short Bs[2][32][128];
    int tid = threadIdx.x;
    int lane = tid & 31, wave = tid >> 5;
    int wm = wave >> 2, wn = wave & 3;
    long m0 = (long)blockIdx.y * 64;
    long n0 = (long)blockIdx.x * 128;

    FragC acc[2][2];
#pragma unroll
    for (int i = 0; i < 2; ++i)
#pragma unroll
        for (int j = 0; j < 2; ++j) acc[i][j].v = zero8();

    int nk = K / 32;
#if USE_ASYNC_LDS
    auto issue_tiles = [&](int p, int kc) {
        int k0 = kc * 32;
        int r = tid >> 2, sg = (tid & 3) * 8;            // A: 64x32 = 256 x 16B
        async_copy16(A + (m0 + r) * K + k0 + sg, &As[p][r][sg]);
#pragma unroll
        for (int q = 0; q < 2; ++q) {                    // B: 32x128 = 512 x 16B
            int s = tid + q * 256;
            int k = s >> 3, sn = (s & 7) * 16;
            async_copy16(Bm + (long)(k0 + k) * N + n0 + sn, &Bs[p][k][sn]);
        }
    };
    issue_tiles(0, 0);
#endif
    for (int kc = 0; kc < nk; ++kc) {
        int p = kc & 1;
#if USE_ASYNC_LDS
        if (kc + 1 < nk) {
            issue_tiles(p ^ 1, kc + 1);                   // next chunk in flight
            __builtin_amdgcn_s_wait_asynccnt(3);          // only older group must land
        } else {
            __builtin_amdgcn_s_wait_asynccnt(0);
        }
#else
        int k0 = kc * 32;
#pragma unroll
        for (int q = 0; q < 4; ++q) {
            int i = tid + q * 256;
            int r = i >> 4, cp = (i & 15) * 2;
            *(unsigned int*)&As[p][r][cp] = *(const unsigned int*)(A + (m0 + r) * K + k0 + cp);
        }
#pragma unroll
        for (int q = 0; q < 8; ++q) {
            int i = tid + q * 256;
            int k = i >> 6, np = (i & 63) * 2;
            *(unsigned int*)&Bs[p][k][np] = *(const unsigned int*)(Bm + (long)(k0 + k) * N + n0 + np);
        }
        if (kc + 1 < nk) {
            __builtin_prefetch(A + (m0 + (tid & 63)) * K + k0 + 32, 0, 0);
            __builtin_prefetch(Bm + (long)(k0 + 32 + (tid & 31)) * N + n0 + ((tid >> 5) << 4), 0, 0);
        }
#endif
        __syncthreads();
        v16bf a0 = lds_a_frag(&As[p][wm * 32][0], 32);
        v16bf a1 = lds_a_frag(&As[p][wm * 32 + 16][0], 32);
        v16bf b0 = lds_b_frag_kn(&Bs[p][0][wn * 32], 128, 0);
        v16bf b1 = lds_b_frag_kn(&Bs[p][0][wn * 32 + 16], 128, 0);
        acc[0][0].v = wmma_bf16(a0, b0, acc[0][0].v);
        acc[0][1].v = wmma_bf16(a0, b1, acc[0][1].v);
        acc[1][0].v = wmma_bf16(a1, b0, acc[1][0].v);
        acc[1][1].v = wmma_bf16(a1, b1, acc[1][1].v);
        __syncthreads();   // readers done -> buffer p may be refilled next round
    }

    int lh = lane >> 4, nl = lane & 15;
#pragma unroll
    for (int i = 0; i < 2; ++i)
#pragma unroll
        for (int j = 0; j < 2; ++j) {
#pragma unroll
            for (int r = 0; r < 8; ++r) {
                long row = m0 + wm * 32 + i * 16 + r + lh * 8;
                long col = n0 + wn * 32 + j * 16 + nl;
                float v = acc[i][j].f[r];
                if (BIAS) v += bias[col];
                if (GELU) { float x = v; v = 0.5f * x * (1.f + tanhf(0.7978845608f * (x + 0.044715f * x * x * x))); }
                if (RES)  v += res[row * N + col];
                long idx = row * N + col;
                if (OUT_F32)  outF[idx] = v;
                if (OUT_BF16) outH[idx] = f2bf(v);
            }
        }
}

// ---------------- flash attention (bf16 WMMA, f32 softmax) ----------------
// grid (L/128, H, B); 8 waves, each wave: 16 query rows. T chunks of 32 keys,
// double-buffered K/V with staggered async waits (2 copies/thread/chunk).
__global__ __launch_bounds__(256) void attention_kernel(
    const unsigned short* __restrict__ Q, const unsigned short* __restrict__ Kg,
    const unsigned short* __restrict__ Vg, unsigned short* __restrict__ O) {
    int blk = blockIdx.x, h = blockIdx.y, b = blockIdx.z;
    int tid = threadIdx.x, lane = tid & 31, wave = tid >> 5;
    int qbase = blk * 128 + wave * 16;
    int lh = lane >> 4, nl = lane & 15;

    __shared__ unsigned short Ks[2][32][64];  // [t][d]
    __shared__ unsigned short Vs[2][32][64];  // [t][d]
    __shared__ unsigned short Ps[8][16][32];  // per-wave P staging

    const unsigned short* Qb = Q + ((long)b * LD + qbase) * DD + h * HSD;
    v16bf aq0 = glb_a_frag(Qb, DD, 0);
    v16bf aq1 = glb_a_frag(Qb, DD, 32);

    float mrow[8], lrow[8];
    FragC acc[4];
#pragma unroll
    for (int r = 0; r < 8; ++r) { mrow[r] = -3.0e38f; lrow[r] = 0.f; }
#pragma unroll
    for (int nt = 0; nt < 4; ++nt) acc[nt].v = zero8();

    const unsigned short* Kb = Kg + (long)b * TD * DD + h * HSD;
    const unsigned short* Vb = Vg + (long)b * TD * DD + h * HSD;
    const int nchunk = TD / 32;

#if USE_ASYNC_LDS
    auto issue_chunk = [&](int p, int tc) {
        int t0 = tc * 32;
        int t = tid >> 3, sg = (tid & 7) * 8;  // 32x64 = 256 x 16B per tile
        async_copy16(Kb + (long)(t0 + t) * DD + sg, &Ks[p][t][sg]);
        async_copy16(Vb + (long)(t0 + t) * DD + sg, &Vs[p][t][sg]);
    };
    issue_chunk(0, 0);
#endif
    for (int tc = 0; tc < nchunk; ++tc) {
        int p = tc & 1;
#if USE_ASYNC_LDS
        if (tc + 1 < nchunk) {
            issue_chunk(p ^ 1, tc + 1);
            __builtin_amdgcn_s_wait_asynccnt(2);
        } else {
            __builtin_amdgcn_s_wait_asynccnt(0);
        }
#else
        int t0 = tc * 32;
#pragma unroll
        for (int q = 0; q < 4; ++q) {
            int i = tid + q * 256;
            int t = i >> 5, dp = (i & 31) * 2;
            *(unsigned int*)&Ks[p][t][dp] = *(const unsigned int*)(Kb + (long)(t0 + t) * DD + dp);
        }
#pragma unroll
        for (int q = 0; q < 4; ++q) {
            int i = tid + q * 256;
            int t = i >> 5, dp = (i & 31) * 2;
            *(unsigned int*)&Vs[p][t][dp] = *(const unsigned int*)(Vb + (long)(t0 + t) * DD + dp);
        }
#endif
        __syncthreads();

        // scores: S[16 x 32] as two 16x16 tiles, k-dim = HS=64 (two wmma each)
        v8f s0 = zero8(), s1 = zero8();
        s0 = wmma_bf16(aq0, lds_b_frag_nk(&Ks[p][0][0],  64, 0),  s0);
        s0 = wmma_bf16(aq1, lds_b_frag_nk(&Ks[p][0][0],  64, 32), s0);
        s1 = wmma_bf16(aq0, lds_b_frag_nk(&Ks[p][16][0], 64, 0),  s1);
        s1 = wmma_bf16(aq1, lds_b_frag_nk(&Ks[p][16][0], 64, 32), s1);
        FragC S0, S1; S0.v = s0; S1.v = s1;

        // online softmax, C-layout: VGPR r holds row (r + 8*lh), lanes = key col
#pragma unroll
        for (int r = 0; r < 8; ++r) {
            float a = S0.f[r] * 0.125f;     // 1/sqrt(64)
            float c = S1.f[r] * 0.125f;
            float mx = fmaxf(a, c);
#pragma unroll
            for (int o = 1; o < 16; o <<= 1) mx = fmaxf(mx, __shfl_xor(mx, o, 32));
            float mn = fmaxf(mrow[r], mx);
            float fac = __expf(mrow[r] - mn);
            mrow[r] = mn;
            float p0 = __expf(a - mn), p1 = __expf(c - mn);
            float ps = p0 + p1;
#pragma unroll
            for (int o = 1; o < 16; o <<= 1) ps += __shfl_xor(ps, o, 32);
            lrow[r] = lrow[r] * fac + ps;
#pragma unroll
            for (int nt = 0; nt < 4; ++nt) acc[nt].f[r] *= fac;
            Ps[wave][r + 8 * lh][nl]      = f2bf(p0);
            Ps[wave][r + 8 * lh][nl + 16] = f2bf(p1);
        }
        // P (16x32) @ V (32x64): one A frag, four strided B frags (per-wave LDS)
        v16bf ap = lds_a_frag(&Ps[wave][0][0], 32);
#pragma unroll
        for (int nt = 0; nt < 4; ++nt)
            acc[nt].v = wmma_bf16(ap, lds_b_frag_kn(&Vs[p][0][nt * 16], 64, 0), acc[nt].v);
        __syncthreads();   // readers done -> buffer p may be refilled next round
    }

    unsigned short* Ob = O + ((long)b * LD + qbase) * DD + h * HSD;
#pragma unroll
    for (int nt = 0; nt < 4; ++nt)
#pragma unroll
        for (int r = 0; r < 8; ++r) {
            float v = acc[nt].f[r] / lrow[r];
            Ob[(long)(r + 8 * lh) * DD + nt * 16 + nl] = f2bf(v);
        }
}

// ---------------- host side ----------------
extern "C" void kernel_launch(void* const* d_in, const int* in_sizes, int n_in,
                              void* d_out, int out_size, void* d_ws, size_t ws_size,
                              hipStream_t stream) {
    const float* grid_in = (const float*)d_in[0];
    const float* qpos    = (const float*)d_in[1];
    const float* Wq      = (const float*)d_in[2];
    const float* Wk      = (const float*)d_in[3];
    const float* Wv      = (const float*)d_in[4];
    const float* Wo      = (const float*)d_in[5];
    const float* lgg     = (const float*)d_in[6];
    const float* lgb     = (const float*)d_in[7];
    const float* lqg     = (const float*)d_in[8];
    const float* lqb     = (const float*)d_in[9];
    const float* lmg     = (const float*)d_in[10];
    const float* lmb     = (const float*)d_in[11];
    const float* W1      = (const float*)d_in[12];
    const float* b1      = (const float*)d_in[13];
    const float* W2      = (const float*)d_in[14];
    const float* b2      = (const float*)d_in[15];
    float* out = (float*)d_out;

    const long BT = (long)BD * TD;   // 4608
    const long BL = (long)BD * LD;   // 8192

    size_t off = 0;
    auto alloc = [&](size_t bytes) {
        void* p = (char*)d_ws + off;
        off = (off + bytes + 255) & ~(size_t)255;
        return p;
    };
    unsigned short* gnorm = (unsigned short*)alloc(BT * DD * 2);
    unsigned short* qnorm = (unsigned short*)alloc(BL * POSD * 2);
    unsigned short* wqh   = (unsigned short*)alloc((size_t)POSD * DD * 2);
    unsigned short* wkh   = (unsigned short*)alloc((size_t)DD * DD * 2);
    unsigned short* wvh   = (unsigned short*)alloc((size_t)DD * DD * 2);
    unsigned short* woh   = (unsigned short*)alloc((size_t)DD * DD * 2);
    unsigned short* w1h   = (unsigned short*)alloc((size_t)DD * DFFD * 2);
    unsigned short* w2h   = (unsigned short*)alloc((size_t)DFFD * DD * 2);
    unsigned short* Kh    = (unsigned short*)alloc(BT * DD * 2);
    unsigned short* Vh    = (unsigned short*)alloc(BT * DD * 2);
    float*          Qf    = (float*)alloc(BL * DD * 4);
    unsigned short* Qh    = (unsigned short*)alloc(BL * DD * 2);
    unsigned short* attn  = (unsigned short*)alloc(BL * DD * 2);
    float*          xf    = (float*)alloc(BL * DD * 4);
    unsigned short* xn    = (unsigned short*)alloc(BL * DD * 2);
    unsigned short* hh    = (unsigned short*)alloc(BL * DFFD * 2);

    auto cvt = [&](const float* s, unsigned short* d, long n) {
        cvt_f32_bf16<<<dim3((unsigned)((n + 255) / 256)), 256, 0, stream>>>(s, d, (int)n);
    };
    cvt(Wq, wqh, (long)POSD * DD);
    cvt(Wk, wkh, (long)DD * DD);
    cvt(Wv, wvh, (long)DD * DD);
    cvt(Wo, woh, (long)DD * DD);
    cvt(W1, w1h, (long)DD * DFFD);
    cvt(W2, w2h, (long)DFFD * DD);

    layernorm_bf16<<<dim3((unsigned)BT), 256, 0, stream>>>(grid_in, lgg, lgb, gnorm, DD);
    layernorm_bf16<<<dim3((unsigned)BL), 256, 0, stream>>>(qpos,    lqg, lqb, qnorm, POSD);

    // K = gnorm @ Wk ; V = gnorm @ Wv   (4608 x 1024, K=1024)
    gemm_bf16<false,false,false,false,true><<<dim3(DD/128, (unsigned)(BT/64)), 256, 0, stream>>>(
        gnorm, wkh, nullptr, nullptr, nullptr, Kh, (int)BT, DD, DD);
    gemm_bf16<false,false,false,false,true><<<dim3(DD/128, (unsigned)(BT/64)), 256, 0, stream>>>(
        gnorm, wvh, nullptr, nullptr, nullptr, Vh, (int)BT, DD, DD);
    // Q = qnorm @ Wq (8192 x 1024, K=512), dual f32 + bf16 output
    gemm_bf16<false,false,false,true,true><<<dim3(DD/128, (unsigned)(BL/64)), 256, 0, stream>>>(
        qnorm, wqh, nullptr, nullptr, Qf, Qh, (int)BL, DD, POSD);

    attention_kernel<<<dim3(LD/128, HD, BD), 256, 0, stream>>>(Qh, Kh, Vh, attn);

    // x = Q + attn @ Wo
    gemm_bf16<false,false,true,true,false><<<dim3(DD/128, (unsigned)(BL/64)), 256, 0, stream>>>(
        attn, woh, nullptr, Qf, xf, nullptr, (int)BL, DD, DD);

    layernorm_bf16<<<dim3((unsigned)BL), 256, 0, stream>>>(xf, lmg, lmb, xn, DD);

    // h = gelu(xn @ W1 + b1)
    gemm_bf16<true,true,false,false,true><<<dim3(DFFD/128, (unsigned)(BL/64)), 256, 0, stream>>>(
        xn, w1h, b1, nullptr, nullptr, hh, (int)BL, DFFD, DD);
    // out = x + h @ W2 + b2
    gemm_bf16<true,false,true,true,false><<<dim3(DD/128, (unsigned)(BL/64)), 256, 0, stream>>>(
        hh, w2h, b2, xf, out, nullptr, (int)BL, DD, DFFD);

    (void)in_sizes; (void)n_in; (void)out_size; (void)ws_size;
}
